// MultiheadSelfAttention_16698832847445
// MI455X (gfx1250) — matrix operand order repaired and checked by reading, not compiled, not run on hardware
//
#include <hip/hip_runtime.h>

// ---------------------------------------------------------------------------
// Problem constants (B=8, S=1024, E=512, H=8, D=64)
// ---------------------------------------------------------------------------
#define BB   8
#define SS   1024
#define EE   512
#define HH   8
#define DD   64
#define MM   (BB * SS)           // 8192 rows for the projection GEMMs
#define NROW (BB * HH * SS)      // 65536 attention rows

typedef __attribute__((ext_vector_type(16))) __bf16 v16bf;
typedef __attribute__((ext_vector_type(8)))  __bf16 v8bf;
typedef __attribute__((ext_vector_type(4)))  __bf16 v4bf;
typedef __attribute__((ext_vector_type(8)))  float  v8f;

static __device__ __forceinline__ v8f wmma_bf16(v16bf a, v16bf b, v8f c) {
    // D = A(16x32 bf16) x B(32x16 bf16) + C(16x16 f32)
    return __builtin_amdgcn_wmma_f32_16x16x32_bf16(
        false, a, false, b, (short)0, c, false, false);
}

// A fragment (16x32): lanes 0-15: row=lane, K in {kb+0..7, kb+16..23};
// lanes 16-31: row=lane-16, K in {kb+8..15, kb+24..31}.   (ISA 7.12.2)
static __device__ __forceinline__ v16bf ld_a_frag(const __bf16* t, int rowBase,
                                                  int stride, int kb, int lane) {
    int r  = rowBase + (lane & 15);
    int ko = kb + ((lane & 16) ? 8 : 0);
    v8bf lo = *reinterpret_cast<const v8bf*>(t + r * stride + ko);
    v8bf hi = *reinterpret_cast<const v8bf*>(t + r * stride + ko + 16);
    return __builtin_shufflevector(lo, hi, 0,1,2,3,4,5,6,7,8,9,10,11,12,13,14,15);
}

// B fragment (32x16), tile stored [n][k]: lanes 0-15: col=lane, K in {kb+0..7, kb+16..23};
// lanes 16-31: col=lane-16, K in {kb+8..15, kb+24..31}.
static __device__ __forceinline__ v16bf ld_b_frag(const __bf16* t, int colBase,
                                                  int stride, int kb, int lane) {
    int c  = colBase + (lane & 15);
    int ko = kb + ((lane & 16) ? 8 : 0);
    v8bf lo = *reinterpret_cast<const v8bf*>(t + c * stride + ko);
    v8bf hi = *reinterpret_cast<const v8bf*>(t + c * stride + ko + 16);
    return __builtin_shufflevector(lo, hi, 0,1,2,3,4,5,6,7,8,9,10,11,12,13,14,15);
}

// ---------------------------------------------------------------------------
// K0: weight prep: Wt[n][k] (bf16) = W[k][n] (f32), for Wq/Wk/Wv/Wo.
// ---------------------------------------------------------------------------
__global__ __launch_bounds__(256)
void wprep_kernel(const float* __restrict__ Wq, const float* __restrict__ Wk,
                  const float* __restrict__ Wv, const float* __restrict__ Wo,
                  __bf16* __restrict__ Wtq, __bf16* __restrict__ Wtk,
                  __bf16* __restrict__ Wtv, __bf16* __restrict__ Wto) {
    const int which = blockIdx.z;
    const float* W = (which == 0) ? Wq : (which == 1) ? Wk : (which == 2) ? Wv : Wo;
    __bf16* Wt     = (which == 0) ? Wtq : (which == 1) ? Wtk : (which == 2) ? Wtv : Wto;

    const int k0 = blockIdx.x * 64, n0 = blockIdx.y * 64;
    const int t = threadIdx.x;
    __shared__ __align__(16) __bf16 T[64 * 80];   // [n][k], stride 80 (160B, 16B-mult)
#pragma unroll
    for (int i = 0; i < 16; ++i) {
        int e = i * 256 + t;
        int r = e >> 6, c = e & 63;               // r = k row, c = n col
        T[c * 80 + r] = (__bf16)W[(size_t)(k0 + r) * EE + n0 + c];
    }
    __syncthreads();
#pragma unroll
    for (int i = 0; i < 2; ++i) {
        int e = i * 256 + t;
        int rr = e >> 3, c8 = (e & 7) * 8;
        *reinterpret_cast<v8bf*>(Wt + (size_t)(n0 + rr) * EE + k0 + c8) =
            *reinterpret_cast<const v8bf*>(T + rr * 80 + c8);
    }
}

// ---------------------------------------------------------------------------
// K1: fused QKV projection. Q,K -> [b,h,s,d] bf16; V -> transposed [b,h,d,s].
// ---------------------------------------------------------------------------
__global__ __launch_bounds__(256)
void qkv_proj_kernel(const float* __restrict__ x,
                     const __bf16* __restrict__ Wtq, const __bf16* __restrict__ Wtk,
                     const __bf16* __restrict__ Wtv,
                     const float* __restrict__ bq, const float* __restrict__ bk,
                     const float* __restrict__ bv,
                     __bf16* __restrict__ Qb, __bf16* __restrict__ Kb,
                     __bf16* __restrict__ Vtg) {
    const int which = blockIdx.z;
    const __bf16* Wt  = (which == 0) ? Wtq : (which == 1) ? Wtk : Wtv;
    const float* bias = (which == 0) ? bq : (which == 1) ? bk : bv;
    __bf16* out       = (which == 0) ? Qb : (which == 1) ? Kb : Vtg;

    const int m0 = blockIdx.x * 64;
    const int n0 = blockIdx.y * 64;
    const int t = threadIdx.x, wave = t >> 5, lane = t & 31;

    __shared__ __align__(16) __bf16 At[64 * 32];   // x tile [m][k]
    __shared__ __align__(16) __bf16 Bt[64 * 32];   // W tile [n][k]

    const int mOff = (wave & 3) * 16;
    const int nOff0 = (wave >> 2) * 32, nOff1 = nOff0 + 16;
    v8f acc0 = {}, acc1 = {};

    for (int kk = 0; kk < EE; kk += 32) {
#pragma unroll
        for (int i = 0; i < 2; ++i) {             // A: f32 -> bf16, float4 chunks
            int e = i * 256 + t;
            int r = e >> 3, c4 = (e & 7) * 4;
            float4 v = *reinterpret_cast<const float4*>(
                x + (size_t)(m0 + r) * EE + kk + c4);
            v4bf p;
            p[0] = (__bf16)v.x; p[1] = (__bf16)v.y;
            p[2] = (__bf16)v.z; p[3] = (__bf16)v.w;
            *reinterpret_cast<v4bf*>(At + r * 32 + c4) = p;
        }
        {                                          // B: straight b128 copy
            int nr = t >> 2, c8 = (t & 3) * 8;
            *reinterpret_cast<v8bf*>(Bt + nr * 32 + c8) =
                *reinterpret_cast<const v8bf*>(Wt + (size_t)(n0 + nr) * EE + kk + c8);
        }
        if (kk + 32 < EE)
            __builtin_prefetch(x + (size_t)(m0 + (t & 63)) * EE + kk + 32, 0, 1);
        __syncthreads();
        v16bf a  = ld_a_frag(At, mOff, 32, 0, lane);
        v16bf b0 = ld_b_frag(Bt, nOff0, 32, 0, lane);
        v16bf b1 = ld_b_frag(Bt, nOff1, 32, 0, lane);
        acc0 = wmma_bf16(a, b0, acc0);
        acc1 = wmma_bf16(a, b1, acc1);
        __syncthreads();
    }

    const int nl = lane & 15, mb = (lane >> 4) * 8;
#pragma unroll
    for (int s = 0; s < 2; ++s) {
        const int nG = n0 + (s ? nOff1 : nOff0) + nl;
        const float bsv = bias[nG];
        const int h = nG >> 6, d = nG & 63;
#pragma unroll
        for (int j = 0; j < 8; ++j) {
            float v = (s ? acc1[j] : acc0[j]) + bsv;
            int mG = m0 + mOff + mb + j;
            int bIdx = mG >> 10, srow = mG & 1023;
            size_t idx = (which == 2)
                ? ((size_t)(bIdx * HH + h) * DD + d) * SS + srow      // V: [b,h,d,s]
                : ((size_t)(bIdx * HH + h) * SS + srow) * DD + d;     // Q,K: [b,h,s,d]
            out[idx] = (__bf16)v;
        }
    }
}

// ---------------------------------------------------------------------------
// K2: per-row squared norms of Q and K (from the bf16 values used in QK^T).
// ---------------------------------------------------------------------------
__global__ __launch_bounds__(256)
void norms_kernel(const __bf16* __restrict__ Qb, const __bf16* __restrict__ Kb,
                  float* __restrict__ qn, float* __restrict__ kn) {
    int i = blockIdx.x * 256 + threadIdx.x;
    if (i >= NROW) return;
    const __bf16* q = Qb + (size_t)i * DD;
    const __bf16* k = Kb + (size_t)i * DD;
    float sq = 0.f, sk = 0.f;
#pragma unroll
    for (int j8 = 0; j8 < 8; ++j8) {
        v8bf qv = *reinterpret_cast<const v8bf*>(q + j8 * 8);
        v8bf kv = *reinterpret_cast<const v8bf*>(k + j8 * 8);
#pragma unroll
        for (int j = 0; j < 8; ++j) {
            float a = (float)qv[j]; sq += a * a;
            float b = (float)kv[j]; sk += b * b;
        }
    }
    qn[i] = sq; kn[i] = sk;
}

// ---------------------------------------------------------------------------
// K3: per-row stats via recomputed QK^T.  Q tile resident in LDS; two passes
// over K tiles with per-lane register accumulation; shuffle+LDS combine.
// ---------------------------------------------------------------------------
__global__ __launch_bounds__(256)
void qk_stats_kernel(const __bf16* __restrict__ Qb, const __bf16* __restrict__ Kb,
                     const float* __restrict__ qn, const float* __restrict__ kn,
                     float* __restrict__ rmax, float* __restrict__ esum,
                     float* __restrict__ rsum) {
    const int bh = blockIdx.y;
    const int q0 = blockIdx.x * 64;
    const int t = threadIdx.x, wave = t >> 5, lane = t & 31;

    __shared__ __align__(16) __bf16 Qt[64 * 64];
    __shared__ __align__(16) __bf16 Kt[64 * 64];
    __shared__ float redM[2][64], redE[2][64], redR[2][64];

    {   // Q tile: rows q0..q0+63 are 4096 contiguous bf16 (row stride == 64)
        const v8bf* src = reinterpret_cast<const v8bf*>(Qb + ((size_t)bh * SS + q0) * DD);
        v8bf* dst = reinterpret_cast<v8bf*>(Qt);
        dst[t] = src[t];
        dst[t + 256] = src[t + 256];
    }

    const int mOff = (wave & 3) * 16;
    const int nOff0 = (wave >> 2) * 32, nOff1 = nOff0 + 16;
    const int nl = lane & 15, mb = (lane >> 4) * 8;

    // ---- pass 1: per-row max of qk/8 -------------------------------------
    float runM[8];
#pragma unroll
    for (int j = 0; j < 8; ++j) runM[j] = -1e30f;

    for (int kt = 0; kt < 16; ++kt) {
        __syncthreads();
        const v8bf* src =
            reinterpret_cast<const v8bf*>(Kb + ((size_t)bh * SS + kt * 64) * DD);
        v8bf* dst = reinterpret_cast<v8bf*>(Kt);
        dst[t] = src[t];
        dst[t + 256] = src[t + 256];
        __syncthreads();
        v8f a0 = {}, a1 = {};
#pragma unroll
        for (int kb = 0; kb < DD; kb += 32) {
            v16bf a  = ld_a_frag(Qt, mOff, 64, kb, lane);
            v16bf b0 = ld_b_frag(Kt, nOff0, 64, kb, lane);
            v16bf b1 = ld_b_frag(Kt, nOff1, 64, kb, lane);
            a0 = wmma_bf16(a, b0, a0);
            a1 = wmma_bf16(a, b1, a1);
        }
#pragma unroll
        for (int j = 0; j < 8; ++j)
            runM[j] = fmaxf(runM[j], fmaxf(a0[j], a1[j]) * 0.125f);
    }
#pragma unroll
    for (int j = 0; j < 8; ++j) {
        float m = runM[j];
#pragma unroll
        for (int msk = 1; msk < 16; msk <<= 1) m = fmaxf(m, __shfl_xor(m, msk, 32));
        runM[j] = m;
    }
    if (nl == 0) {          // lanes 0 and 16 of every wave
#pragma unroll
        for (int j = 0; j < 8; ++j) redM[wave >> 2][mOff + mb + j] = runM[j];
    }
    __syncthreads();

    float bmj[8], qnr[8], sigj[8];
#pragma unroll
    for (int j = 0; j < 8; ++j) {
        int r = mOff + mb + j;
        bmj[j] = fmaxf(redM[0][r], redM[1][r]);
        float q = qn[(size_t)bh * SS + q0 + r];
        qnr[j] = q;
        sigj[j] = fminf(fmaxf(q, 1e-8f), 1e4f);
    }

    // ---- pass 2: exp-sum and RBF-sum -------------------------------------
    float runE[8] = {}, runR[8] = {};
    for (int kt = 0; kt < 16; ++kt) {
        __syncthreads();
        const v8bf* src =
            reinterpret_cast<const v8bf*>(Kb + ((size_t)bh * SS + kt * 64) * DD);
        v8bf* dst = reinterpret_cast<v8bf*>(Kt);
        dst[t] = src[t];
        dst[t + 256] = src[t + 256];
        __syncthreads();
        v8f a0 = {}, a1 = {};
#pragma unroll
        for (int kb = 0; kb < DD; kb += 32) {
            v16bf a  = ld_a_frag(Qt, mOff, 64, kb, lane);
            v16bf b0 = ld_b_frag(Kt, nOff0, 64, kb, lane);
            v16bf b1 = ld_b_frag(Kt, nOff1, 64, kb, lane);
            a0 = wmma_bf16(a, b0, a0);
            a1 = wmma_bf16(a, b1, a1);
        }
        float kn0 = kn[(size_t)bh * SS + kt * 64 + nOff0 + nl];
        float kn1 = kn[(size_t)bh * SS + kt * 64 + nOff1 + nl];
#pragma unroll
        for (int j = 0; j < 8; ++j) {
            float s0 = a0[j], s1 = a1[j];
            runE[j] += __expf(s0 * 0.125f - bmj[j]) + __expf(s1 * 0.125f - bmj[j]);
            float d0 = fmaxf(qnr[j] + kn0 - 2.f * s0, 0.f);
            float d1 = fmaxf(qnr[j] + kn1 - 2.f * s1, 0.f);
            runR[j] += __expf(-d0 / sigj[j]) + __expf(-d1 / sigj[j]);
        }
    }
#pragma unroll
    for (int j = 0; j < 8; ++j) {
        float e = runE[j], r = runR[j];
#pragma unroll
        for (int msk = 1; msk < 16; msk <<= 1) {
            e += __shfl_xor(e, msk, 32);
            r += __shfl_xor(r, msk, 32);
        }
        runE[j] = e; runR[j] = r;
    }
    if (nl == 0) {
#pragma unroll
        for (int j = 0; j < 8; ++j) {
            redE[wave >> 2][mOff + mb + j] = runE[j];
            redR[wave >> 2][mOff + mb + j] = runR[j];
        }
    }
    __syncthreads();
    if (t < 64) {
        size_t ri = (size_t)bh * SS + q0 + t;
        rmax[ri] = fmaxf(redM[0][t], redM[1][t]);
        esum[ri] = redE[0][t] + redE[1][t];
        rsum[ri] = redR[0][t] + redR[1][t];
    }
}

// ---------------------------------------------------------------------------
// K4: fused QK^T recompute + blend; writes final attn (fp32) exactly once.
// ---------------------------------------------------------------------------
__global__ __launch_bounds__(256)
void qk_blend_kernel(const __bf16* __restrict__ Qb, const __bf16* __restrict__ Kb,
                     const float* __restrict__ qn, const float* __restrict__ kn,
                     const float* __restrict__ rmax, const float* __restrict__ esum,
                     const float* __restrict__ rsum, const float* __restrict__ pond,
                     float* __restrict__ attn) {
    const int bh = blockIdx.z;
    const int q0 = blockIdx.x * 64, k0 = blockIdx.y * 64;
    const int t = threadIdx.x, wave = t >> 5, lane = t & 31;

    __shared__ __align__(16) __bf16 Qt[64 * 64];
    __shared__ __align__(16) __bf16 Kt[64 * 64];
    {
        const v8bf* sq = reinterpret_cast<const v8bf*>(Qb + ((size_t)bh * SS + q0) * DD);
        const v8bf* sk = reinterpret_cast<const v8bf*>(Kb + ((size_t)bh * SS + k0) * DD);
        v8bf* dq = reinterpret_cast<v8bf*>(Qt);
        v8bf* dk = reinterpret_cast<v8bf*>(Kt);
        dq[t] = sq[t]; dq[t + 256] = sq[t + 256];
        dk[t] = sk[t]; dk[t + 256] = sk[t + 256];
    }
    __syncthreads();

    const int mOff = (wave & 3) * 16;
    const int nOff0 = (wave >> 2) * 32, nOff1 = nOff0 + 16;
    v8f acc0 = {}, acc1 = {};
#pragma unroll
    for (int kb = 0; kb < DD; kb += 32) {
        v16bf a  = ld_a_frag(Qt, mOff, 64, kb, lane);
        v16bf b0 = ld_b_frag(Kt, nOff0, 64, kb, lane);
        v16bf b1 = ld_b_frag(Kt, nOff1, 64, kb, lane);
        acc0 = wmma_bf16(a, b0, acc0);
        acc1 = wmma_bf16(a, b1, acc1);
    }

    const int nl = lane & 15, mb = (lane >> 4) * 8;
    const float sv = 1.f / (1.f + __expf(-pond[0]));
    const float p0 = 1.f - sv, p1 = sv;
    const float pin = 1.f / (p0 + p1 + 1e-7f);

    float bm[8], iE[8], iR[8], qnr[8], sigj[8];
#pragma unroll
    for (int j = 0; j < 8; ++j) {
        size_t ri = (size_t)bh * SS + q0 + mOff + mb + j;
        bm[j] = rmax[ri];
        iE[j] = 1.f / esum[ri];
        iR[j] = 1.f / fmaxf(rsum[ri], 1e-8f);
        float q = qn[ri];
        qnr[j] = q;
        sigj[j] = fminf(fmaxf(q, 1e-8f), 1e4f);
    }
    const float kn0 = kn[(size_t)bh * SS + k0 + nOff0 + nl];
    const float kn1 = kn[(size_t)bh * SS + k0 + nOff1 + nl];

#pragma unroll
    for (int s = 0; s < 2; ++s) {
        int kG = k0 + (s ? nOff1 : nOff0) + nl;
        float knc = s ? kn1 : kn0;
#pragma unroll
        for (int j = 0; j < 8; ++j) {
            float qv = s ? acc1[j] : acc0[j];
            float sm = __expf(qv * 0.125f - bm[j]) * iE[j];
            float d = fmaxf(qnr[j] + knc - 2.f * qv, 0.f);
            float rbf = __expf(-d / sigj[j]) * iR[j];
            if (!(rbf == rbf) || rbf > 3.0e38f || rbf < -3.0e38f) rbf = 0.f;
            size_t ri = (size_t)bh * SS + q0 + mOff + mb + j;
            attn[ri * SS + kG] = (p0 * sm + p1 * rbf) * pin;
        }
    }
}

// ---------------------------------------------------------------------------
// K5: ctx[b,s,h*64+d] (bf16) = attn @ V   (V pre-transposed [b,h,d,s])
// ---------------------------------------------------------------------------
__global__ __launch_bounds__(256)
void av_kernel(const float* __restrict__ attn, const __bf16* __restrict__ Vtg,
               __bf16* __restrict__ ctx) {
    const int q0 = blockIdx.x * 64;
    const int bh = blockIdx.y;
    const int t = threadIdx.x, wave = t >> 5, lane = t & 31;

    __shared__ __align__(16) __bf16 At[64 * 32];   // attn tile [q][k]
    __shared__ __align__(16) __bf16 Vt[64 * 32];   // V tile [d][k]

    const int mOff = (wave & 3) * 16;
    const int nOff0 = (wave >> 2) * 32, nOff1 = nOff0 + 16;
    v8f acc0 = {}, acc1 = {};

    for (int kk = 0; kk < SS; kk += 32) {
#pragma unroll
        for (int i = 0; i < 2; ++i) {              // attn f32 -> bf16
            int e = i * 256 + t;
            int r = e >> 3, c4 = (e & 7) * 4;
            float4 v = *reinterpret_cast<const float4*>(
                attn + ((size_t)bh * SS + q0 + r) * SS + kk + c4);
            v4bf p;
            p[0] = (__bf16)v.x; p[1] = (__bf16)v.y;
            p[2] = (__bf16)v.z; p[3] = (__bf16)v.w;
            *reinterpret_cast<v4bf*>(At + r * 32 + c4) = p;
        }
        {                                          // V: straight b128 copy
            int dl = t >> 2, c8 = (t & 3) * 8;
            *reinterpret_cast<v8bf*>(Vt + dl * 32 + c8) =
                *reinterpret_cast<const v8bf*>(Vtg + ((size_t)bh * DD + dl) * SS + kk + c8);
        }
        __syncthreads();
        v16bf a  = ld_a_frag(At, mOff, 32, 0, lane);
        v16bf b0 = ld_b_frag(Vt, nOff0, 32, 0, lane);
        v16bf b1 = ld_b_frag(Vt, nOff1, 32, 0, lane);
        acc0 = wmma_bf16(a, b0, acc0);
        acc1 = wmma_bf16(a, b1, acc1);
        __syncthreads();
    }

    const int nl = lane & 15, mb = (lane >> 4) * 8;
    const int b = bh >> 3, h = bh & 7;
#pragma unroll
    for (int s = 0; s < 2; ++s) {
        int dG = (s ? nOff1 : nOff0) + nl;
#pragma unroll
        for (int j = 0; j < 8; ++j) {
            int qG = q0 + mOff + mb + j;
            ctx[((size_t)(b * SS + qG)) * EE + h * DD + dG] =
                (__bf16)(s ? acc1[j] : acc0[j]);
        }
    }
}

// ---------------------------------------------------------------------------
// K6: out = ctx @ Wo + bo   (fp32 output)
// ---------------------------------------------------------------------------
__global__ __launch_bounds__(256)
void oproj_kernel(const __bf16* __restrict__ ctx, const __bf16* __restrict__ Wto,
                  const float* __restrict__ bo, float* __restrict__ out) {
    const int m0 = blockIdx.x * 64;
    const int n0 = blockIdx.y * 64;
    const int t = threadIdx.x, wave = t >> 5, lane = t & 31;

    __shared__ __align__(16) __bf16 At[64 * 32];
    __shared__ __align__(16) __bf16 Bt[64 * 32];

    const int mOff = (wave & 3) * 16;
    const int nOff0 = (wave >> 2) * 32, nOff1 = nOff0 + 16;
    v8f acc0 = {}, acc1 = {};

    for (int kk = 0; kk < EE; kk += 32) {
        {
            int r = t >> 2, c8 = (t & 3) * 8;
            *reinterpret_cast<v8bf*>(At + r * 32 + c8) =
                *reinterpret_cast<const v8bf*>(ctx + (size_t)(m0 + r) * EE + kk + c8);
            *reinterpret_cast<v8bf*>(Bt + r * 32 + c8) =
                *reinterpret_cast<const v8bf*>(Wto + (size_t)(n0 + r) * EE + kk + c8);
        }
        __syncthreads();
        v16bf a  = ld_a_frag(At, mOff, 32, 0, lane);
        v16bf b0 = ld_b_frag(Bt, nOff0, 32, 0, lane);
        v16bf b1 = ld_b_frag(Bt, nOff1, 32, 0, lane);
        acc0 = wmma_bf16(a, b0, acc0);
        acc1 = wmma_bf16(a, b1, acc1);
        __syncthreads();
    }

    const int nl = lane & 15, mb = (lane >> 4) * 8;
#pragma unroll
    for (int s = 0; s < 2; ++s) {
        int nG = n0 + (s ? nOff1 : nOff0) + nl;
        float bs = bo[nG];
#pragma unroll
        for (int j = 0; j < 8; ++j) {
            int mG = m0 + mOff + mb + j;
            out[(size_t)mG * EE + nG] = (s ? acc1[j] : acc0[j]) + bs;
        }
    }
}

// ---------------------------------------------------------------------------
extern "C" void kernel_launch(void* const* d_in, const int* in_sizes, int n_in,
                              void* d_out, int out_size, void* d_ws,
                              size_t ws_size, hipStream_t stream) {
    const float* x    = (const float*)d_in[0];
    const float* Wq   = (const float*)d_in[1];
    const float* bq   = (const float*)d_in[2];
    const float* Wk   = (const float*)d_in[3];
    const float* bk   = (const float*)d_in[4];
    const float* Wv   = (const float*)d_in[5];
    const float* bv   = (const float*)d_in[6];
    const float* Wo   = (const float*)d_in[7];
    const float* bo   = (const float*)d_in[8];
    const float* pond = (const float*)d_in[9];

    float* out  = (float*)d_out;                       // [B,S,E]
    float* attn = out + (size_t)BB * SS * EE;          // [B,H,S,S]

    // Workspace carve-up (~35.5 MB)
    char* ws = (char*)d_ws;
    size_t off = 0;
    __bf16* Qb  = (__bf16*)(ws + off); off += (size_t)NROW * DD * 2;   // 8 MB
    __bf16* Kb  = (__bf16*)(ws + off); off += (size_t)NROW * DD * 2;   // 8 MB
    __bf16* Vtg = (__bf16*)(ws + off); off += (size_t)NROW * DD * 2;   // 8 MB [b,h,d,s]
    __bf16* ctx = (__bf16*)(ws + off); off += (size_t)MM * EE * 2;     // 8 MB
    __bf16* Wtq = (__bf16*)(ws + off); off += (size_t)EE * EE * 2;     // 512 KB each
    __bf16* Wtk = (__bf16*)(ws + off); off += (size_t)EE * EE * 2;
    __bf16* Wtv = (__bf16*)(ws + off); off += (size_t)EE * EE * 2;
    __bf16* Wto = (__bf16*)(ws + off); off += (size_t)EE * EE * 2;
    float* qn   = (float*)(ws + off);  off += (size_t)NROW * 4;
    float* kn   = (float*)(ws + off);  off += (size_t)NROW * 4;
    float* rmax = (float*)(ws + off);  off += (size_t)NROW * 4;
    float* esum = (float*)(ws + off);  off += (size_t)NROW * 4;
    float* rsum = (float*)(ws + off);  off += (size_t)NROW * 4;

    wprep_kernel<<<dim3(EE / 64, EE / 64, 4), 256, 0, stream>>>(
        Wq, Wk, Wv, Wo, Wtq, Wtk, Wtv, Wto);
    qkv_proj_kernel<<<dim3(MM / 64, EE / 64, 3), 256, 0, stream>>>(
        x, Wtq, Wtk, Wtv, bq, bk, bv, Qb, Kb, Vtg);
    norms_kernel<<<dim3(NROW / 256), 256, 0, stream>>>(Qb, Kb, qn, kn);
    qk_stats_kernel<<<dim3(SS / 64, BB * HH), 256, 0, stream>>>(
        Qb, Kb, qn, kn, rmax, esum, rsum);
    qk_blend_kernel<<<dim3(SS / 64, SS / 64, BB * HH), 256, 0, stream>>>(
        Qb, Kb, qn, kn, rmax, esum, rsum, pond, attn);
    av_kernel<<<dim3(SS / 64, BB * HH), 256, 0, stream>>>(attn, Vtg, ctx);
    oproj_kernel<<<dim3(MM / 64, EE / 64), 256, 0, stream>>>(ctx, Wto, bo, out);
}
